// GAAPModelB_893353198383
// MI455X (gfx1250) — compile-verified, matrix operand order checked by reading, not run on hardware
//
#include <hip/hip_runtime.h>
#include <hip/hip_bf16.h>
#include <stdint.h>

// ---------------------------------------------------------------------------
// GAAP GNN pipeline for gfx1250 (MI455X), v3.
// bf16 end-to-end dataflow; WMMA operands come from LDS-staged activation
// tiles (bank-conflict-free via +16B row padding) and L2-resident transposed
// bf16 weights.  Inner loop: 2x ds_load_b128 + 2x global_load_b128 per WMMA.
// ---------------------------------------------------------------------------

#define NN   100000
#define EE   800000
#define IND  128
#define HD   256
#define GC   512
#define PAD  8        // +8 bf16 = +16B LDS row padding -> conflict-free frags

typedef __attribute__((ext_vector_type(16))) __bf16 v16bf;
typedef __attribute__((ext_vector_type(8)))  __bf16 v8bf;
typedef __attribute__((ext_vector_type(8)))  float  v8f;

// f32 -> bf16 RNE (epilogue / elementwise only)
__device__ __forceinline__ __bf16 f2bf(float f) {
    unsigned u = __builtin_bit_cast(unsigned, f);
    unsigned r = u + 0x7FFFu + ((u >> 16) & 1u);
    unsigned short h = (unsigned short)(r >> 16);
    return __builtin_bit_cast(__bf16, h);
}
__device__ __forceinline__ float bf2f(__bf16 b) {
    unsigned u = (unsigned)__builtin_bit_cast(unsigned short, b) << 16;
    return __builtin_bit_cast(float, u);
}

// 16x32 fragment from a row-major bf16 matrix (works for global or LDS).
// ISA layout: lane r (0..15): row r, K 0..7 / 16..23 ; lane r+16: K 8..15 / 24..31.
__device__ __forceinline__ v16bf load_frag(const __bf16* base, int ld, int r0, int k0) {
    int lane = threadIdx.x & 31;
    int half = lane >> 4;
    int r    = lane & 15;
    const __bf16* p = base + (size_t)(r0 + r) * ld + (k0 + half * 8);
    v8bf lo = *(const v8bf*)p;
    v8bf hi = *(const v8bf*)(p + 16);
    return __builtin_shufflevector(lo, hi, 0,1,2,3,4,5,6,7,8,9,10,11,12,13,14,15);
}

#define WMMA_BF16(c, a, b) \
    __builtin_amdgcn_wmma_f32_16x16x32_bf16(false, (a), false, (b), (short)0, (c), false, false)

// ---------------------------------------------------------------------------
// Fused GEMM: C = act(A1[Nr x K1] @ W1t^T (+ A2 @ W2t^T) + bias), bf16 out.
// A tiles are contiguous 16-row slabs -> cooperatively staged in padded LDS.
// Weights pre-transposed (Wt[M x K]) so B fragments are contiguous b128s.
// ---------------------------------------------------------------------------
__global__ void gemm2_wmma(const __bf16* __restrict__ A1, int K1,
                           const __bf16* __restrict__ W1t, int ldw1,
                           const __bf16* __restrict__ A2, int K2,
                           const __bf16* __restrict__ W2t, int ldw2,
                           const float* __restrict__ bias,
                           __bf16* __restrict__ C, int M, int ldc,
                           int relu, int transpose)
{
    extern __shared__ __bf16 smem[];
    __bf16* s1 = smem;                       // [16][K1+PAD]
    __bf16* s2 = smem + 16 * (K1 + PAD);     // [16][K2+PAD]
    int row0 = blockIdx.x * 16;

    // cooperative LDS staging (16B chunks, coalesced; rows are contiguous)
    {
        int ck1 = K1 >> 3;
        for (int c = threadIdx.x; c < 16 * ck1; c += 256) {
            int row = c / ck1, kc = c - row * ck1;
            *(v8bf*)(s1 + row * (K1 + PAD) + kc * 8) =
                *(const v8bf*)(A1 + (size_t)(row0 + row) * K1 + kc * 8);
        }
        if (A2) {
            int ck2 = K2 >> 3;
            for (int c = threadIdx.x; c < 16 * ck2; c += 256) {
                int row = c / ck2, kc = c - row * ck2;
                *(v8bf*)(s2 + row * (K2 + PAD) + kc * 8) =
                    *(const v8bf*)(A2 + (size_t)(row0 + row) * K2 + kc * 8);
            }
        }
    }
    __syncthreads();                         // all waves reach this barrier

    int wave    = threadIdx.x >> 5;
    int colTile = blockIdx.y * 8 + wave;
    if (colTile * 16 >= M) return;           // wave-uniform; no barriers after
    int col0 = colTile * 16;

    v8f c = {};
    for (int k0 = 0; k0 < K1; k0 += 32) {
        v16bf a = load_frag(s1, K1 + PAD, 0, k0);      // ds_load_b128 x2
        v16bf b = load_frag(W1t, ldw1, col0, k0);      // global_load_b128 x2
        c = WMMA_BF16(c, a, b);
    }
    if (A2) {
        for (int k0 = 0; k0 < K2; k0 += 32) {
            v16bf a = load_frag(s2, K2 + PAD, 0, k0);
            v16bf b = load_frag(W2t, ldw2, col0, k0);
            c = WMMA_BF16(c, a, b);
        }
    }

    int lane = threadIdx.x & 31;
    int n    = lane & 15;
    int half = lane >> 4;
    float bv = bias ? bias[col0 + n] : 0.0f;
    if (!transpose) {
#pragma unroll
        for (int r = 0; r < 8; ++r) {
            float v = c[r] + bv;
            if (relu) v = fmaxf(v, 0.0f);
            C[(size_t)(row0 + half * 8 + r) * ldc + col0 + n] = f2bf(v);
        }
    } else {
#pragma unroll
        for (int r = 0; r < 8; ++r) {
            float v = c[r] + bv;
            if (relu) v = fmaxf(v, 0.0f);
            C[(size_t)(col0 + n) * ldc + row0 + half * 8 + r] = f2bf(v);
        }
    }
}

// ---------------------------------------------------------------------------
// One-time converts
// ---------------------------------------------------------------------------
__global__ void cvt_f32_bf16_kernel(const float* __restrict__ in, __bf16* __restrict__ out,
                                    long long n) {
    long long i = (long long)blockIdx.x * blockDim.x + threadIdx.x;
    if (i < n) out[i] = f2bf(in[i]);
}

// Wt[M x K] = (bf16) W[K x M]^T
__global__ void transpose_cvt_kernel(const float* __restrict__ W, __bf16* __restrict__ Wt,
                                     int K, int M) {
    int idx = blockIdx.x * blockDim.x + threadIdx.x;
    if (idx >= K * M) return;
    int k = idx / M, m = idx % M;
    Wt[(size_t)m * K + k] = f2bf(W[idx]);
}

// ---------------------------------------------------------------------------
// Edge kernels: degree count + bf16-gather / f32-atomic scatter-add.
// ---------------------------------------------------------------------------
__global__ void edge_count_kernel(const int* __restrict__ dst, float* __restrict__ cnt) {
    size_t e = (size_t)blockIdx.x * blockDim.x + threadIdx.x;
    if (e < EE) atomicAdd(&cnt[dst[e]], 1.0f);
}

__global__ void edge_scatter_kernel(const int* __restrict__ src, const int* __restrict__ dst,
                                    const __bf16* __restrict__ h, float* __restrict__ agg,
                                    int logd)
{
    size_t idx = (size_t)blockIdx.x * blockDim.x + threadIdx.x;
    size_t e   = idx >> logd;
    if (e >= EE) return;
    int d = 1 << logd;
    int f = (int)(idx & (d - 1));
    int s = src[e], t = dst[e];
    atomicAdd(&agg[(size_t)t * d + f], bf2f(h[(size_t)s * d + f]));
}

__global__ void mean_div_kernel(const float* __restrict__ agg, const float* __restrict__ cnt,
                                __bf16* __restrict__ mean, int logd) {
    size_t idx = (size_t)blockIdx.x * blockDim.x + threadIdx.x;
    if (idx >= ((size_t)NN << logd)) return;
    float c = cnt[idx >> logd];
    mean[idx] = f2bf(agg[idx] / fmaxf(c, 1.0f));
}

// ---------------------------------------------------------------------------
// Global attention per 16-node tile.  Q tile staged in padded LDS (read once
// instead of 32x); Kc[512,256] / Vt[256,512] bf16 stay L2-resident.
// ---------------------------------------------------------------------------
__global__ void attn_kernel(const __bf16* __restrict__ Q, const __bf16* __restrict__ Kc,
                            const __bf16* __restrict__ Vt, __bf16* __restrict__ g)
{
    __shared__ float  S[16][GC];
    __shared__ __bf16 Pb[16][GC + PAD];
    __shared__ __bf16 Qs[16][HD + PAD];
    __shared__ float  red[16][16];

    int lane = threadIdx.x & 31;
    int wave = threadIdx.x >> 5;
    int row0 = blockIdx.x * 16;
    int n    = lane & 15;
    int half = lane >> 4;

    // stage Q tile (16x256 contiguous) into padded LDS
    for (int c = threadIdx.x; c < 16 * (HD / 8); c += 256) {
        int row = c >> 5, kc = c & 31;
        *(v8bf*)(&Qs[row][kc * 8]) = *(const v8bf*)(Q + (size_t)(row0 + row) * HD + kc * 8);
    }
    __syncthreads();

    // Phase 1: S = (Q @ K^T) / sqrt(H)  -- 32 col tiles of 16, 4 per wave
#pragma unroll
    for (int t = 0; t < 4; ++t) {
        int col0 = (wave * 4 + t) * 16;
        v8f c = {};
        for (int k0 = 0; k0 < HD; k0 += 32) {
            v16bf a = load_frag(&Qs[0][0], HD + PAD, 0, k0);   // ds fragments
            v16bf b = load_frag(Kc, HD, col0, k0);             // Kc row = cache idx
            c = WMMA_BF16(c, a, b);
        }
#pragma unroll
        for (int r = 0; r < 8; ++r)
            S[half * 8 + r][col0 + n] = c[r] * 0.0625f;        // 1/sqrt(256)
    }
    __syncthreads();

    // Phase 2: row softmax (16 threads per row), emit bf16 P
    int row = threadIdx.x >> 4;
    int sub = threadIdx.x & 15;
    float mx = -3.402823466e38f;
    for (int j = sub; j < GC; j += 16) mx = fmaxf(mx, S[row][j]);
    red[row][sub] = mx;
    __syncthreads();
    float rmax = red[row][0];
#pragma unroll
    for (int i = 1; i < 16; ++i) rmax = fmaxf(rmax, red[row][i]);
    __syncthreads();
    float sm = 0.0f;
    for (int j = sub; j < GC; j += 16) {
        float e = __expf(S[row][j] - rmax);
        S[row][j] = e;
        sm += e;
    }
    red[row][sub] = sm;
    __syncthreads();
    float rsum = 0.0f;
#pragma unroll
    for (int i = 0; i < 16; ++i) rsum += red[row][i];
    float inv = 1.0f / rsum;
    for (int j = sub; j < GC; j += 16) Pb[row][j] = f2bf(S[row][j] * inv);
    __syncthreads();

    // Phase 3: g = P @ V  -- 16 col tiles of 16, 2 per wave, K = 512 from LDS
#pragma unroll
    for (int t = 0; t < 2; ++t) {
        int col0 = (wave * 2 + t) * 16;
        v8f c = {};
        for (int k0 = 0; k0 < GC; k0 += 32) {
            v16bf a = load_frag(&Pb[0][0], GC + PAD, 0, k0);   // ds fragments
            v16bf b = load_frag(Vt, GC, col0, k0);             // Vt row = out col
            c = WMMA_BF16(c, a, b);
        }
#pragma unroll
        for (int r = 0; r < 8; ++r)
            g[(size_t)(row0 + half * 8 + r) * HD + col0 + n] = f2bf(c[r]);
    }
}

// ---------------------------------------------------------------------------
// Final tiny GEMM (M=2): VALU dot products, bandwidth bound on m1.
// ---------------------------------------------------------------------------
__global__ void final_mlp_kernel(const __bf16* __restrict__ m1, const float* __restrict__ Wm2,
                                 const float* __restrict__ bm2, float* __restrict__ out)
{
    int row = blockIdx.x * blockDim.x + threadIdx.x;
    if (row >= NN) return;
    const __bf16* p = m1 + (size_t)row * HD;
    float a0 = bm2[0], a1 = bm2[1];
#pragma unroll 4
    for (int j = 0; j < HD; ++j) {
        float v = bf2f(p[j]);
        a0 = fmaf(v, Wm2[j * 2 + 0], a0);
        a1 = fmaf(v, Wm2[j * 2 + 1], a1);
    }
    out[(size_t)row * 2 + 0] = a0;
    out[(size_t)row * 2 + 1] = a1;
}

// ---------------------------------------------------------------------------
// Host orchestration
// ---------------------------------------------------------------------------
extern "C" void kernel_launch(void* const* d_in, const int* in_sizes, int n_in,
                              void* d_out, int out_size, void* d_ws, size_t ws_size,
                              hipStream_t stream)
{
    (void)in_sizes; (void)n_in; (void)out_size; (void)ws_size;

    const float* x    = (const float*)d_in[0];
    const int*   eidx = (const int*)  d_in[1];
    const float* gcch = (const float*)d_in[2];
    const float* Wp   = (const float*)d_in[3];
    const float* bp   = (const float*)d_in[4];
    const float* W1l  = (const float*)d_in[5];
    const float* b1l  = (const float*)d_in[6];
    const float* W1r  = (const float*)d_in[7];
    const float* W2l  = (const float*)d_in[8];
    const float* b2l  = (const float*)d_in[9];
    const float* W2r  = (const float*)d_in[10];
    const float* Wq   = (const float*)d_in[11];
    const float* bq   = (const float*)d_in[12];
    const float* Wk   = (const float*)d_in[13];
    const float* bk   = (const float*)d_in[14];
    const float* Wv   = (const float*)d_in[15];
    const float* bv   = (const float*)d_in[16];
    const float* Wm1  = (const float*)d_in[17];   // [512,256]
    const float* bm1  = (const float*)d_in[18];
    const float* Wm2  = (const float*)d_in[19];   // [256,2]
    const float* bm2  = (const float*)d_in[20];
    float*       out  = (float*)d_out;

    const int* esrc = eidx;
    const int* edst = eidx + EE;

    // ---- workspace carve-out (all sizes 16B multiples; d_ws is 256B aligned)
    char* w = (char*)d_ws;
    auto allocF = [&](size_t n) { float*  p = (float*) w; w += n * 4; return p; };
    auto allocB = [&](size_t n) { __bf16* p = (__bf16*)w; w += n * 2; return p; };

    float*  cnt    = allocF(NN);
    float*  agg64  = allocF((size_t)NN * 64);
    float*  agg256 = allocF((size_t)NN * HD);

    __bf16* xb     = allocB((size_t)NN * IND);
    __bf16* xemb   = allocB((size_t)NN * 64);
    __bf16* mean64 = allocB((size_t)NN * 64);
    __bf16* h1     = allocB((size_t)NN * HD);
    __bf16* mn256  = allocB((size_t)NN * HD);
    __bf16* h2     = allocB((size_t)NN * HD);
    __bf16* Qb     = allocB((size_t)NN * HD);    // Q, reused as m1 after attention
    __bf16* gb     = allocB((size_t)NN * HD);
    __bf16* gcb    = allocB((size_t)GC * HD);
    __bf16* Kc     = allocB((size_t)GC * HD);
    __bf16* Vt     = allocB((size_t)HD * GC);    // V-proj transposed [256x512]

    __bf16* Wpt    = allocB((size_t)64  * IND);
    __bf16* W1lt   = allocB((size_t)HD  * 64);
    __bf16* W1rt   = allocB((size_t)HD  * 64);
    __bf16* W2lt   = allocB((size_t)HD  * HD);
    __bf16* W2rt   = allocB((size_t)HD  * HD);
    __bf16* Wqt    = allocB((size_t)HD  * HD);
    __bf16* Wkt    = allocB((size_t)HD  * HD);
    __bf16* Wvt    = allocB((size_t)HD  * HD);
    __bf16* Wm1t   = allocB((size_t)HD  * 2 * HD);

    const int ROWT = NN / 16;   // 6250 exact
    dim3 blk(256);
    auto g1 = [](size_t n) { return dim3((unsigned)((n + 255) / 256)); };
    auto smemSz = [](int K1, int K2) {
        return (size_t)2 * (16 * (K1 + PAD) + (K2 ? 16 * (K2 + PAD) : 0));
    };

    // ---- zero f32 accumulators (graph-capture safe)
    hipMemsetAsync(cnt,    0, sizeof(float) * NN, stream);
    hipMemsetAsync(agg64,  0, sizeof(float) * (size_t)NN * 64, stream);
    hipMemsetAsync(agg256, 0, sizeof(float) * (size_t)NN * HD, stream);

    // ---- one-time converts
    cvt_f32_bf16_kernel<<<g1((size_t)NN * IND), blk, 0, stream>>>(x,    xb,  (long long)NN * IND);
    cvt_f32_bf16_kernel<<<g1((size_t)GC * HD),  blk, 0, stream>>>(gcch, gcb, (long long)GC * HD);
    transpose_cvt_kernel<<<g1((size_t)IND * 64),    blk, 0, stream>>>(Wp,  Wpt,  IND, 64);
    transpose_cvt_kernel<<<g1((size_t)64 * HD),     blk, 0, stream>>>(W1l, W1lt, 64,  HD);
    transpose_cvt_kernel<<<g1((size_t)64 * HD),     blk, 0, stream>>>(W1r, W1rt, 64,  HD);
    transpose_cvt_kernel<<<g1((size_t)HD * HD),     blk, 0, stream>>>(W2l, W2lt, HD,  HD);
    transpose_cvt_kernel<<<g1((size_t)HD * HD),     blk, 0, stream>>>(W2r, W2rt, HD,  HD);
    transpose_cvt_kernel<<<g1((size_t)HD * HD),     blk, 0, stream>>>(Wq,  Wqt,  HD,  HD);
    transpose_cvt_kernel<<<g1((size_t)HD * HD),     blk, 0, stream>>>(Wk,  Wkt,  HD,  HD);
    transpose_cvt_kernel<<<g1((size_t)HD * HD),     blk, 0, stream>>>(Wv,  Wvt,  HD,  HD);
    transpose_cvt_kernel<<<g1((size_t)2 * HD * HD), blk, 0, stream>>>(Wm1, Wm1t, 2 * HD, HD);

    // 1. x_emb = relu(x @ Wp + bp)                        [N,64]
    gemm2_wmma<<<dim3(ROWT, 1), blk, smemSz(IND, 0), stream>>>(
        xb, IND, Wpt, IND, nullptr, 0, nullptr, 0, bp, xemb, 64, 64, 1, 0);

    // degrees (shared by both SAGE layers)
    edge_count_kernel<<<g1(EE), blk, 0, stream>>>(edst, cnt);

    // 2. SAGE1 aggregate
    edge_scatter_kernel<<<g1((size_t)EE * 64), blk, 0, stream>>>(esrc, edst, xemb, agg64, 6);
    mean_div_kernel<<<g1((size_t)NN * 64), blk, 0, stream>>>(agg64, cnt, mean64, 6);

    // 3. h1 = relu(mean64 @ W1l + b1l + x_emb @ W1r)      [N,256]
    gemm2_wmma<<<dim3(ROWT, 2), blk, smemSz(64, 64), stream>>>(
        mean64, 64, W1lt, 64, xemb, 64, W1rt, 64, b1l, h1, HD, HD, 1, 0);

    // 4. SAGE2 aggregate
    edge_scatter_kernel<<<g1((size_t)EE * HD), blk, 0, stream>>>(esrc, edst, h1, agg256, 8);
    mean_div_kernel<<<g1((size_t)NN * HD), blk, 0, stream>>>(agg256, cnt, mn256, 8);

    // 5. h2 = mean256 @ W2l + b2l + h1 @ W2r              [N,256]
    gemm2_wmma<<<dim3(ROWT, 2), blk, smemSz(HD, HD), stream>>>(
        mn256, HD, W2lt, HD, h1, HD, W2rt, HD, b2l, h2, HD, HD, 0, 0);

    // 6. Q = h2 @ Wq + bq                                 [N,256]
    gemm2_wmma<<<dim3(ROWT, 2), blk, smemSz(HD, 0), stream>>>(
        h2, HD, Wqt, HD, nullptr, 0, nullptr, 0, bq, Qb, HD, HD, 0, 0);

    // 7. K/V projections of the cache; V stored transposed
    gemm2_wmma<<<dim3(GC / 16, 2), blk, smemSz(HD, 0), stream>>>(
        gcb, HD, Wkt, HD, nullptr, 0, nullptr, 0, bk, Kc, HD, HD, 0, 0);
    gemm2_wmma<<<dim3(GC / 16, 2), blk, smemSz(HD, 0), stream>>>(
        gcb, HD, Wvt, HD, nullptr, 0, nullptr, 0, bv, Vt, HD, GC, 0, 1);

    // 8. g = softmax(Q K^T / 16) V                        [N,256]
    attn_kernel<<<dim3(ROWT), blk, 0, stream>>>(Qb, Kc, Vt, gb);

    // 9. m1 = relu([h2|g] @ Wm1 + bm1): split K halves of Wm1t [256 x 512]
    gemm2_wmma<<<dim3(ROWT, 2), blk, smemSz(HD, HD), stream>>>(
        h2, HD, Wm1t, 2 * HD, gb, HD, Wm1t + HD, 2 * HD,
        bm1, Qb /*reuse as m1*/, HD, HD, 1, 0);

    // 10. out = m1 @ Wm2 + bm2                            [N,2]
    final_mlp_kernel<<<g1(NN), blk, 0, stream>>>(Qb, Wm2, bm2, out);
}
